// GSoP_31894427140282
// MI455X (gfx1250) — compile-verified, matrix operand order checked by reading, not run on hardware
//
#include <hip/hip_runtime.h>
#include <hip/hip_bf16.h>

typedef __bf16 bf16_t;
typedef __attribute__((ext_vector_type(16))) __bf16 v16bf;
typedef __attribute__((ext_vector_type(8)))  __bf16 v8bf;
typedef __attribute__((ext_vector_type(8)))  float  v8f;
typedef __attribute__((ext_vector_type(4)))  unsigned int v4u;
typedef __attribute__((ext_vector_type(8)))  int v8i;
typedef __attribute__((ext_vector_type(4)))  int v4i;

#define B_   64
#define CIN  2048
#define MM   196            // H*W
#define C_   128
#define MP   224            // padded M for cov GEMM (7*32)
#define NPAD 208            // 13*16 padded cols in DR gemm
#define EPS  1e-5f

#if __has_builtin(__builtin_amdgcn_tensor_load_to_lds)
#define HAVE_TDM 1
#endif

// Load a 16x32 bf16 WMMA operand fragment from LDS.
// Storage is row-major [16 rows][32 k] (64B per row). Per the CDNA5 16-bit
// A-matrix layout: lanes 0-15 hold row (lane), K in {0..7, 16..23};
// lanes 16-31 hold row (lane-16), K in {8..15, 24..31}.
// sel = 0 for lanes<16, 16 for lanes>=16; chunks at +sel and +32+sel.
__device__ __forceinline__ v16bf ld_frag(const char* rowbase, int sel) {
    v8bf lo = *(const v8bf*)(rowbase + sel);
    v8bf hi = *(const v8bf*)(rowbase + 32 + sel);
    v16bf r;
#pragma unroll
    for (int i = 0; i < 8; ++i) { r[i] = lo[i]; r[i + 8] = hi[i]; }
    return r;
}

__device__ __forceinline__ v8f vzero8() {
    v8f z = {0.f, 0.f, 0.f, 0.f, 0.f, 0.f, 0.f, 0.f};
    return z;
}

// ---------------------------------------------------------------------------
// Kernel 1: Y[b] = relu(BN(W_dr @ relu(X[b]) + b_dr))  -> bf16 [C_ x MP]
// One block per batch; 8 waves; K-chunked (32) bf16 WMMA GEMM from LDS.
// ---------------------------------------------------------------------------
__global__ __launch_bounds__(256)
void gsop_dr_gemm(const float* __restrict__ x, const float* __restrict__ w_dr,
                  const float* __restrict__ b_dr,
                  const float* __restrict__ g1, const float* __restrict__ bb1,
                  const float* __restrict__ m1, const float* __restrict__ v1,
                  bf16_t* __restrict__ Y) {
    __shared__ bf16_t aLds[C_ * 32];      // W tile:  [row c 0..127][k 0..31]
    __shared__ bf16_t bLds[NPAD * 32];    // Xt tile: [col m 0..207][k 0..31]

    const int b    = blockIdx.x;
    const int tid  = threadIdx.x;
    const int w    = tid >> 5;            // wave id 0..7 -> C row-tile
    const int lane = tid & 31;
    const int sel  = (lane >> 4) << 4;    // 0 / 16
    const int rsub = lane & 15;
    const int hi8  = (lane >> 4) << 3;    // 0 / 8

    v8f acc[13];
#pragma unroll
    for (int n = 0; n < 13; ++n) acc[n] = vzero8();

    const float* xb = x + (size_t)b * CIN * MM;

    for (int kc = 0; kc < CIN / 32; ++kc) {
        __syncthreads();
        // Stage A: W_dr[:, kc*32 .. kc*32+32) as bf16 (coalesced reads).
#pragma unroll
        for (int i = 0; i < 16; ++i) {
            int e = tid + i * 256;        // 4096 = 128*32
            int row = e >> 5, kk = e & 31;
            aLds[e] = (bf16_t)w_dr[row * CIN + kc * 32 + kk];
        }
        // Stage Bt: bLds[m][kk] = relu(x[b][kc*32+kk][m]), zero for m>=196.
#pragma unroll
        for (int i = 0; i < 26; ++i) {
            int e = tid + i * 256;        // 6656 = 208*32
            int kk = e / NPAD, mcol = e % NPAD;
            float vv = 0.f;
            if (mcol < MM) {
                float xv = xb[(kc * 32 + kk) * MM + mcol];
                vv = xv > 0.f ? xv : 0.f;
            }
            bLds[mcol * 32 + kk] = (bf16_t)vv;
        }
        // Prefetch next chunk of x (32 rows x 196 floats): 256 cachelines.
        if (kc + 1 < CIN / 32) {
            const float* nx = xb + (size_t)((kc + 1) * 32 + (tid >> 3)) * MM +
                              ((tid & 7) << 5);
            __builtin_prefetch(nx, 0, 1);
        }
        __syncthreads();

        const char* abase = (const char*)aLds + (w * 16 + rsub) * 64;
        v16bf afrag = ld_frag(abase, sel);
#pragma unroll
        for (int n = 0; n < 13; ++n) {
            const char* bbase = (const char*)bLds + (n * 16 + rsub) * 64;
            v16bf bfrag = ld_frag(bbase, sel);
            acc[n] = __builtin_amdgcn_wmma_f32_16x16x32_bf16(
                false, afrag, false, bfrag, (short)0, acc[n], false, false);
        }
    }

    // Epilogue: fold conv bias + BN1 + relu.  y = relu(acc*s + t)
    float s8[8], t8[8];
#pragma unroll
    for (int vi = 0; vi < 8; ++vi) {
        int c = w * 16 + hi8 + vi;
        float s = g1[c] * rsqrtf(v1[c] + EPS);
        s8[vi] = s;
        t8[vi] = (b_dr[c] - m1[c]) * s + bb1[c];
    }
    bf16_t* Yb = Y + (size_t)b * C_ * MP;
#pragma unroll
    for (int n = 0; n < 13; ++n) {
        int col = n * 16 + rsub;
#pragma unroll
        for (int vi = 0; vi < 8; ++vi) {
            int c = w * 16 + hi8 + vi;
            float y = acc[n][vi] * s8[vi] + t8[vi];
            y = y > 0.f ? y : 0.f;
            Yb[c * MP + col] = (bf16_t)((col < MM) ? y : 0.f);
        }
    }
    // zero the K-pad columns 208..223 so the cov GEMM padding is exact
    if (tid < C_) {
        for (int col = NPAD; col < MP; ++col) Yb[tid * MP + col] = (bf16_t)0.f;
    }
}

// ---------------------------------------------------------------------------
// Kernel 2: cov = BN_row(Y Y^T / M - mu mu^T); z = grouped row-conv;
//           att = sigmoid(w_fc @ z + b_fc).  One block per batch.
// Y[b] -> LDS via the Tensor Data Mover when the builtin is available.
// ---------------------------------------------------------------------------
__global__ __launch_bounds__(256)
void gsop_cov_att(const bf16_t* __restrict__ Y,
                  const float* __restrict__ g2, const float* __restrict__ bb2,
                  const float* __restrict__ m2, const float* __restrict__ v2,
                  const float* __restrict__ w_row, const float* __restrict__ b_row,
                  const float* __restrict__ w_fc, const float* __restrict__ b_fc,
                  float* __restrict__ att) {
    // [0, 57344): Y (bf16 128x224), reused afterwards as cov (bf16 128x128)
    // [57344, 57856): mu (f32 x128);  [57856, 59904): z (f32 x512)
    __shared__ __align__(16) unsigned char smem[C_ * MP * 2 + C_ * 4 + 512 * 4];
    bf16_t* Ylds = (bf16_t*)smem;
    float*  muL  = (float*)(smem + C_ * MP * 2);
    float*  zL   = (float*)(smem + C_ * MP * 2 + C_ * 4);

    const int b    = blockIdx.x;
    const int tid  = threadIdx.x;
    const int w    = tid >> 5;
    const int lane = tid & 31;
    const int sel  = (lane >> 4) << 4;
    const int rsub = lane & 15;
    const int hi8  = (lane >> 4) << 3;

#ifdef HAVE_TDM
    // TDM: DMA Y[b] (2D tensor, 2-byte elems, 128 rows x 224 cols) into LDS.
    if (tid < 32) {   // one wave issues the descriptor (EXEC is ignored)
        unsigned long long ga =
            (unsigned long long)(size_t)(Y + (size_t)b * C_ * MP);
        unsigned int lds = (unsigned int)(size_t)&smem[0];
        // D# group 0: count=1 | lds_addr | global_addr[56:0] | type=2
        v4u g0;
        g0.x = 1u;
        g0.y = lds;
        g0.z = (unsigned int)(ga & 0xFFFFFFFFull);
        g0.w = (unsigned int)((ga >> 32) & 0x01FFFFFFull) | (2u << 30);
        // D# group 1: data_size=2B; tensor 224x128; tile 224x128; stride0=224
        v8i g1;
        g1[0] = (1 << 16);            // wg_mask=0, data_size=1 (2 bytes)
        g1[1] = (MP << 16);           // tensor_dim0 = 224 (low 16 in [63:48])
        g1[2] = (C_ << 16);           // tensor_dim0 hi=0, tensor_dim1 = 128
        g1[3] = (MP << 16);           // tensor_dim1 hi=0, tile_dim0 = 224
        g1[4] = C_;                   // tile_dim1 = 128, tile_dim2 = 0
        g1[5] = MP;                   // tensor_dim0_stride lo = 224
        g1[6] = ((MP * C_) << 16);    // stride0 hi=0, tensor_dim1_stride lo
        g1[7] = 0;
        v4i gz = {0, 0, 0, 0};
#if __clang_major__ >= 23
        v8i gz8 = {0, 0, 0, 0, 0, 0, 0, 0};
        __builtin_amdgcn_tensor_load_to_lds(g0, g1, gz, gz, gz8, 0);
#else
        __builtin_amdgcn_tensor_load_to_lds(g0, g1, gz, gz, 0);
#endif
        __builtin_amdgcn_s_wait_tensorcnt(0);
    }
    __syncthreads();
#else
    // Fallback: manual 57344B = 3584 x 16B copy.
    {
        const uint4* src = (const uint4*)(Y + (size_t)b * C_ * MP);
        uint4* dst = (uint4*)smem;
#pragma unroll
        for (int i = 0; i < 14; ++i) dst[tid + i * 256] = src[tid + i * 256];
    }
    __syncthreads();
#endif

    // Per-channel means over the spatial dim.
    if (tid < C_) {
        float s = 0.f;
        for (int m = 0; m < MM; ++m) s += (float)Ylds[tid * MP + m];
        muL[tid] = s * (1.f / MM);
    }
    __syncthreads();

    // G = Y * Y^T from LDS (Bt[n][k] == Y[n][k], same addressing as A).
    v8f acc[8];
#pragma unroll
    for (int n = 0; n < 8; ++n) acc[n] = vzero8();
    const char* base = (const char*)Ylds;
    for (int kc = 0; kc < MP / 32; ++kc) {
        v16bf afrag = ld_frag(base + (w * 16 + rsub) * (MP * 2) + kc * 64, sel);
#pragma unroll
        for (int n = 0; n < 8; ++n) {
            v16bf bfrag = ld_frag(base + (n * 16 + rsub) * (MP * 2) + kc * 64, sel);
            acc[n] = __builtin_amdgcn_wmma_f32_16x16x32_bf16(
                false, afrag, false, bfrag, (short)0, acc[n], false, false);
        }
    }

    // Per-row (i) BN2 params + means.
    float mui[8], ss[8], tt[8], mm[8];
#pragma unroll
    for (int vi = 0; vi < 8; ++vi) {
        int i = w * 16 + hi8 + vi;
        mui[vi] = muL[i];
        float s = g2[i] * rsqrtf(v2[i] + EPS);
        ss[vi] = s;
        mm[vi] = m2[i];
        tt[vi] = bb2[i];
    }
    __syncthreads();   // all waves done reading Ylds; safe to overwrite

    bf16_t* covL = (bf16_t*)smem;
    const float invM = 1.f / (float)MM;
#pragma unroll
    for (int n = 0; n < 8; ++n) {
        int j = n * 16 + rsub;
        float muj = muL[j];
#pragma unroll
        for (int vi = 0; vi < 8; ++vi) {
            int i = w * 16 + hi8 + vi;
            float covv = acc[n][vi] * invM - mui[vi] * muj;
            covL[i * C_ + j] = (bf16_t)((covv - mm[vi]) * ss[vi] + tt[vi]);
        }
    }
    __syncthreads();

    // z[c*4+k] = b_row[c*4+k] + sum_j cov[c][j] * w_row[c][k][j]
#pragma unroll
    for (int t = 0; t < 2; ++t) {
        int idx = tid * 2 + t;                       // 0..511
        int c = idx >> 2, k = idx & 3;
        const float* wr = w_row + (size_t)(c * 4 + k) * C_;
        float s = b_row[idx];
        for (int j = 0; j < C_; ++j) s += (float)covL[c * C_ + j] * wr[j];
        zL[idx] = s;
    }
    __syncthreads();

    // att[o] = sigmoid(b_fc[o] + sum_t z[t] * w_fc[o][t])
    float* attb = att + (size_t)b * CIN;
#pragma unroll
    for (int i = 0; i < 8; ++i) {
        int o = tid + i * 256;
        const float* wf = w_fc + (size_t)o * (4 * C_);
        float s = b_fc[o];
        for (int t = 0; t < 4 * C_; ++t) s += zL[t] * wf[t];
        attb[o] = 1.f / (1.f + __expf(-s));
    }
}

// ---------------------------------------------------------------------------
// Kernel 3: out = x * att (broadcast over spatial dim). Bandwidth-bound tail.
// ---------------------------------------------------------------------------
__global__ __launch_bounds__(256)
void gsop_apply(const float* __restrict__ x, const float* __restrict__ att,
                float* __restrict__ out) {
    const size_t total = (size_t)B_ * CIN * MM;
    size_t idx = (size_t)blockIdx.x * blockDim.x + threadIdx.x;
    const size_t stride = (size_t)gridDim.x * blockDim.x;
    for (; idx < total; idx += stride) {
        out[idx] = x[idx] * att[idx / MM];
    }
}

extern "C" void kernel_launch(void* const* d_in, const int* in_sizes, int n_in,
                              void* d_out, int out_size, void* d_ws, size_t ws_size,
                              hipStream_t stream) {
    (void)in_sizes; (void)n_in; (void)out_size; (void)ws_size;
    const float* x     = (const float*)d_in[0];
    const float* w_dr  = (const float*)d_in[1];
    const float* b_dr  = (const float*)d_in[2];
    const float* g1    = (const float*)d_in[3];
    const float* bb1   = (const float*)d_in[4];
    const float* m1    = (const float*)d_in[5];
    const float* v1    = (const float*)d_in[6];
    const float* g2    = (const float*)d_in[7];
    const float* bb2   = (const float*)d_in[8];
    const float* m2    = (const float*)d_in[9];
    const float* v2    = (const float*)d_in[10];
    const float* w_row = (const float*)d_in[11];
    const float* b_row = (const float*)d_in[12];
    const float* w_fc  = (const float*)d_in[13];
    const float* b_fc  = (const float*)d_in[14];

    bf16_t* Yws  = (bf16_t*)d_ws;
    float*  attw = (float*)((char*)d_ws + (size_t)B_ * C_ * MP * sizeof(bf16_t));

    gsop_dr_gemm<<<B_, 256, 0, stream>>>(x, w_dr, b_dr, g1, bb1, m1, v1, Yws);
    gsop_cov_att<<<B_, 256, 0, stream>>>(Yws, g2, bb2, m2, v2, w_row, b_row,
                                         w_fc, b_fc, attw);
    gsop_apply<<<2048, 256, 0, stream>>>(x, attw, (float*)d_out);
}
